// MatchingModule_5918464933937
// MI455X (gfx1250) — matrix operand order repaired and checked by reading, not compiled, not run on hardware
//
#include <hip/hip_runtime.h>
#include <cstddef>

#define Bn 16
#define Nn 1024
#define Hn 1024

typedef __bf16 bf16_t;
typedef bf16_t v16bf  __attribute__((ext_vector_type(16)));
typedef bf16_t bf16x8 __attribute__((ext_vector_type(8)));
typedef float  v8f    __attribute__((ext_vector_type(8)));

#define SHUF16(lo, hi) __builtin_shufflevector(lo, hi, \
    0, 1, 2, 3, 4, 5, 6, 7, 8, 9, 10, 11, 12, 13, 14, 15)

__device__ __forceinline__ unsigned short f2bfu(float f) {
  union { float f; unsigned u; } v; v.f = f;
  unsigned u = v.u;
  unsigned r = (u + 0x7FFFu + ((u >> 16) & 1u)) >> 16;  // round-to-nearest-even
  return (unsigned short)r;
}

// ---------------------------------------------------------------------------
// Kernel 1: f32 -> bf16 conversion of the two attention matrices (A2O uses
// aspect hidden states, O2A uses opinion hidden states since args.T is even).
// Also zeroes the 32-float loss accumulator block.
// ---------------------------------------------------------------------------
__global__ __launch_bounds__(256) void convert_bf16_kernel(
    const float* __restrict__ a0, const float* __restrict__ a1,
    unsigned long long* __restrict__ o0, unsigned long long* __restrict__ o1,
    float* __restrict__ accb) {
  if (blockIdx.x == 0 && threadIdx.x < 32) accb[threadIdx.x] = 0.0f;
  const size_t n4 = (size_t)Bn * Nn * Hn / 4;
  size_t idx = (size_t)blockIdx.x * blockDim.x + threadIdx.x;
  const size_t stride = (size_t)gridDim.x * blockDim.x;
  const float4* s0 = (const float4*)a0;
  const float4* s1 = (const float4*)a1;
  for (; idx < n4; idx += stride) {
    float4 x = s0[idx];
    unsigned long long p = (unsigned long long)f2bfu(x.x)
                         | ((unsigned long long)f2bfu(x.y) << 16)
                         | ((unsigned long long)f2bfu(x.z) << 32)
                         | ((unsigned long long)f2bfu(x.w) << 48);
    o0[idx] = p;
    float4 y = s1[idx];
    unsigned long long q = (unsigned long long)f2bfu(y.x)
                         | ((unsigned long long)f2bfu(y.y) << 16)
                         | ((unsigned long long)f2bfu(y.z) << 32)
                         | ((unsigned long long)f2bfu(y.w) << 48);
    o1[idx] = q;
  }
}

// ---------------------------------------------------------------------------
// Kernel 2: masked row-argmax of X * X^T via bf16 WMMA.
// Grid: (Nn/16, Bn).  Block: 256 threads = 8 wave32 waves.
// Each block handles one (batch, 16-row tile).  Wave w owns j-tiles
// {4w..4w+3, 4w+32..4w+35}; 4 accumulators amortize each A-fragment LDS read
// across 4 v_wmma issues.  The k-loop is software-pipelined (prefetch
// distance 1) so B-fragment global loads overlap the WMMAs of the previous
// k-step instead of forcing s_wait_loadcnt 0 before every WMMA.
// ---------------------------------------------------------------------------
__global__ __launch_bounds__(256) void score_argmax_kernel(
    const bf16_t* __restrict__ X, const int* __restrict__ aspm,
    const int* __restrict__ opim, int* __restrict__ ind) {
  __shared__ __align__(16) bf16_t Atile[16 * Hn];   // 32 KB
  __shared__ unsigned char opiS[Nn];                // 1 KB
  __shared__ float redS[8][16][16];                 // 8 KB
  __shared__ int   redJ[8][16][16];                 // 8 KB

  const int tid  = threadIdx.x;
  const int wid  = tid >> 5;
  const int lane = tid & 31;
  const int cl   = lane & 15;   // column (B/C) or row (A) within half-wave
  const int hl   = lane >> 4;   // half-wave selector
  const int i0   = blockIdx.x * 16;
  const int bb   = blockIdx.y;

  // Stage the 16 contiguous A rows (bf16) into LDS.
  {
    const int4* src = (const int4*)(X + ((size_t)bb * Nn + i0) * Hn);
    int4* dst = (int4*)Atile;
    for (int t = tid; t < (16 * Hn * 2) / 16; t += 256) dst[t] = src[t];
  }
  for (int t = tid; t < Nn; t += 256)
    opiS[t] = (unsigned char)(opim[bb * Nn + t] != 0);
  __syncthreads();

  const float NEG = -__builtin_inff();
  float best[8];
  int   bestj[8];
#pragma unroll
  for (int v = 0; v < 8; ++v) { best[v] = NEG; bestj[v] = 0x7fffffff; }

  // A-fragment source: lane holds row cl, K-chunks [k0+8*hl,+8), [k0+16+8*hl,+8)
  const bf16_t* arow = &Atile[cl * Hn];
  const int ka = 8 * hl;
  // B-fragment: lane holds column cl (= row j0+cl of X), K-chunks [k0+8*hl,+8), [k0+16+8*hl,+8)
  const int kb = 8 * hl;

  for (int jtb = wid * 4; jtb < 64; jtb += 32) {
    // 32-bit element offsets from the uniform base X -> saddr+voffset form.
    unsigned boff[4];
#pragma unroll
    for (int s = 0; s < 4; ++s)
      boff[s] = ((unsigned)(bb * Nn + (jtb + s) * 16 + cl)) * Hn + kb;

    v8f acc[4] = {};

    // ---- prologue: load fragments for k0 = 0 ----
    bf16x8 alo = *(const bf16x8*)(arow + ka);
    bf16x8 ahi = *(const bf16x8*)(arow + 16 + ka);
    bf16x8 blo[4], bhi[4];
#pragma unroll
    for (int s = 0; s < 4; ++s) {
      blo[s] = *(const bf16x8*)(X + boff[s]);
      bhi[s] = *(const bf16x8*)(X + boff[s] + 16);
    }

    // ---- steady state: prefetch k0+32 while issuing WMMAs for k0 ----
    for (int k0 = 0; k0 < Hn - 32; k0 += 32) {
      v16bf av = SHUF16(alo, ahi);
      v16bf bv[4];
#pragma unroll
      for (int s = 0; s < 4; ++s) bv[s] = SHUF16(blo[s], bhi[s]);

      const int kn = k0 + 32;
      alo = *(const bf16x8*)(arow + kn + ka);
      ahi = *(const bf16x8*)(arow + kn + 16 + ka);
#pragma unroll
      for (int s = 0; s < 4; ++s) {
        blo[s] = *(const bf16x8*)(X + boff[s] + kn);
        bhi[s] = *(const bf16x8*)(X + boff[s] + kn + 16);
      }
#pragma unroll
      for (int s = 0; s < 4; ++s)
        acc[s] = __builtin_amdgcn_wmma_f32_16x16x32_bf16(
            false, av, false, bv[s], (short)0, acc[s], false, false);
    }

    // ---- epilogue: last k-step ----
    {
      v16bf av = SHUF16(alo, ahi);
#pragma unroll
      for (int s = 0; s < 4; ++s) {
        v16bf bv = SHUF16(blo[s], bhi[s]);
        acc[s] = __builtin_amdgcn_wmma_f32_16x16x32_bf16(
            false, av, false, bv, (short)0, acc[s], false, false);
      }
    }

    // Per-lane masked argmax update.  argmax(A) == argmax over valid
    // (opi[j] && j != i) of the raw dot product (exp/denominator monotone).
#pragma unroll
    for (int s = 0; s < 4; ++s) {
      const int j = (jtb + s) * 16 + cl;
      const bool jv = opiS[j] != 0;
#pragma unroll
      for (int v = 0; v < 8; ++v) {
        const int gi = i0 + v + 8 * hl;                 // C row for VGPR v
        float val = (jv && (j != gi)) ? acc[s][v] : NEG;
        if (val > best[v] || (val == best[v] && j < bestj[v])) {
          best[v] = val; bestj[v] = j;
        }
      }
    }
  }

#pragma unroll
  for (int v = 0; v < 8; ++v) {
    redS[wid][v + 8 * hl][cl] = best[v];
    redJ[wid][v + 8 * hl][cl] = bestj[v];
  }
  __syncthreads();

  if (tid < 16) {
    float bs = NEG; int bj = 0;
    for (int w = 0; w < 8; ++w)
      for (int c = 0; c < 16; ++c) {
        float s = redS[w][tid][c];
        int j = redJ[w][tid][c];
        if (s > bs || (s == bs && j < bj)) { bs = s; bj = j; }
      }
    const int aspv = aspm[bb * Nn + i0 + tid];
    // all-invalid row (or aspect-masked row) -> argmax of zeros -> 0
    ind[bb * Nn + i0 + tid] = (aspv != 0 && bs > NEG) ? bj : 0;
  }
}

// ---------------------------------------------------------------------------
// Kernel 3: gather + [asp || gathered] @ W + b per direction, mask, average,
// label argmax, weighted-CE partial sums per batch.  Grid (Nn, Bn), block 256.
// ---------------------------------------------------------------------------
__global__ __launch_bounds__(256) void head_kernel(
    const float* __restrict__ asp0, const float* __restrict__ opi0,
    const float* __restrict__ asp1, const float* __restrict__ opi1,
    const int* __restrict__ m0, const int* __restrict__ m1,
    const int* __restrict__ sent,
    const float* __restrict__ W0, const float* __restrict__ bia0,
    const float* __restrict__ W1, const float* __restrict__ bia1,
    const int* __restrict__ ind0, const int* __restrict__ ind1,
    float* __restrict__ out, float* __restrict__ accb) {
  __shared__ float red[6][256];
  __shared__ float lg[6];
  const int tid = threadIdx.x;
  const int i = blockIdx.x;
  const int b = blockIdx.y;
  const int ri = b * Nn + i;
  const size_t r0 = (size_t)ri * Hn;
  const size_t g0 = ((size_t)b * Nn + ind0[ri]) * Hn;
  const size_t g1 = ((size_t)b * Nn + ind1[ri]) * Hn;

  float acc[6] = {0.f, 0.f, 0.f, 0.f, 0.f, 0.f};
  for (int h = tid; h < 2 * Hn; h += 256) {
    const float x0 = (h < Hn) ? asp0[r0 + h] : opi0[g0 + (h - Hn)];
    const float x1 = (h < Hn) ? asp1[r0 + h] : opi1[g1 + (h - Hn)];
#pragma unroll
    for (int k = 0; k < 3; ++k) {
      acc[k]     += x0 * W0[h * 3 + k];
      acc[3 + k] += x1 * W1[h * 3 + k];
    }
  }
#pragma unroll
  for (int k = 0; k < 6; ++k) red[k][tid] = acc[k];
  __syncthreads();
  if (tid < 6) {
    float s = 0.f;
    for (int t = 0; t < 256; ++t) s += red[tid][t];
    lg[tid] = s;
  }
  __syncthreads();

  if (tid == 0) {
    const bool a0 = m0[ri] != 0;
    const bool a1 = m1[ri] != 0;
    float fl[3];
#pragma unroll
    for (int k = 0; k < 3; ++k) {
      const float l0 = a0 ? (lg[k] + bia0[k]) : 0.f;      // (final@W + b) * asp_m
      const float l1 = a1 ? (lg[3 + k] + bia1[k]) : 0.f;
      fl[k] = 0.5f * (l0 + l1);
    }
    const bool valid = a0 || a1;
    int lbl = 0;
    if (fl[1] > fl[lbl]) lbl = 1;   // strict > keeps first max, like jnp.argmax
    if (fl[2] > fl[lbl]) lbl = 2;
    out[ri] = (float)(valid ? lbl : -1);
    out[Bn * Nn + ri * 3 + 0] = fl[0];
    out[Bn * Nn + ri * 3 + 1] = fl[1];
    out[Bn * Nn + ri * 3 + 2] = fl[2];

    const int s = sent[ri];
    const float wt = (s == 0) ? 1.f : ((s == 1) ? 2.f : 4.f);
    const float w = valid ? wt : 0.f;
    const float m = fmaxf(fl[0], fmaxf(fl[1], fl[2]));
    const float lse = m + logf(expf(fl[0] - m) + expf(fl[1] - m) + expf(fl[2] - m));
    const float ce = lse - fl[s];
    atomicAdd(&accb[b], w * ce);
    atomicAdd(&accb[16 + b], w);
  }
}

// ---------------------------------------------------------------------------
// Kernel 4: lossS = sum_b (den_b > 0 ? num_b / den_b : 0)
// ---------------------------------------------------------------------------
__global__ void loss_kernel(const float* __restrict__ accb, float* __restrict__ out) {
  __shared__ float s[16];
  const int t = threadIdx.x;
  if (t < 16) {
    const float d = accb[16 + t];
    s[t] = (d > 0.f) ? accb[t] / d : 0.f;
  }
  __syncthreads();
  if (t == 0) {
    float L = 0.f;
    for (int k = 0; k < 16; ++k) L += s[k];
    out[Bn * Nn * 4] = L;  // after labels (B*N) + logits (B*N*3)
  }
}

// ---------------------------------------------------------------------------
extern "C" void kernel_launch(void* const* d_in, const int* in_sizes, int n_in,
                              void* d_out, int out_size, void* d_ws, size_t ws_size,
                              hipStream_t stream) {
  (void)in_sizes; (void)n_in; (void)out_size; (void)ws_size;
  const float* A2O_asp = (const float*)d_in[0];
  const float* A2O_opi = (const float*)d_in[1];
  const float* O2A_asp = (const float*)d_in[2];
  const float* O2A_opi = (const float*)d_in[3];
  const int* aspm0 = (const int*)d_in[4];
  const int* opim0 = (const int*)d_in[5];
  const int* aspm1 = (const int*)d_in[6];
  const int* opim1 = (const int*)d_in[7];
  const int* sent  = (const int*)d_in[8];
  const float* W0 = (const float*)d_in[9];
  const float* b0 = (const float*)d_in[10];
  const float* W1 = (const float*)d_in[11];
  const float* b1 = (const float*)d_in[12];
  float* out = (float*)d_out;

  char* wsb = (char*)d_ws;
  const size_t XB = (size_t)Bn * Nn * Hn * 2;           // 32 MB per direction
  bf16_t* X0 = (bf16_t*)wsb;                            // A2O attn (aspect) bf16
  bf16_t* X1 = (bf16_t*)(wsb + XB);                     // O2A attn (opinion) bf16
  int* ind0 = (int*)(wsb + 2 * XB);
  int* ind1 = ind0 + Bn * Nn;
  float* accb = (float*)(ind1 + Bn * Nn);               // num[16], den[16]

  convert_bf16_kernel<<<2048, 256, 0, stream>>>(
      A2O_asp, O2A_opi, (unsigned long long*)X0, (unsigned long long*)X1, accb);

  dim3 gs(Nn / 16, Bn);
  score_argmax_kernel<<<gs, 256, 0, stream>>>(X0, aspm0, opim0, ind0);
  score_argmax_kernel<<<gs, 256, 0, stream>>>(X1, aspm1, opim1, ind1);

  dim3 gh(Nn, Bn);
  head_kernel<<<gh, 256, 0, stream>>>(A2O_asp, A2O_opi, O2A_asp, O2A_opi,
                                      aspm0, aspm1, sent, W0, b0, W1, b1,
                                      ind0, ind1, out, accb);
  loss_kernel<<<1, 32, 0, stream>>>(accb, out);
}
